// NormalizeCLAHEBlur_31258771980578
// MI455X (gfx1250) — matrix operand order repaired and checked by reading, not compiled, not run on hardware
//
#include <hip/hip_runtime.h>
#include <hip/hip_bf16.h>
#include <math.h>

typedef __attribute__((ext_vector_type(16))) _Float16 v16h;
typedef __attribute__((ext_vector_type(8)))  float    v8f;

// Problem constants (reference: x = (32,1,1024,1024) f32, y = (32,1024,1024) i32)
#define IMG_B   32
#define IMG_H   1024
#define IMG_W   1024
#define N_TOT   (33554432u)     // 32*1024*1024
#define GH      8
#define GW      8
#define TILE    128             // 1024/8
#define PIX     16384           // 128*128
#define NB      256
#define CLIPVAL 192.0f          // max(3*16384//256, 1)

struct WsHeader {
  unsigned xmin_key, xmax_key;   // min/max of raw x
  unsigned lmin_key, lmax_key;   // min/max of log-adjusted l
  unsigned bmin_key, bmax_key;   // min/max of blurred output
  unsigned pad0, pad1;
  double   sum, sumsq;           // stats of l
  float    mn, inv_range;        // raw-x normalize params
  float    su, ou;               // u = su*l + ou  (standardize + renormalize folded)
  float    pad2[4];
};
// ws layout: [WsHeader (256B-aligned region)] [float lut[2048*256]] (2 MB, L2-resident)

// ---- order-preserving float<->uint keys for atomic min/max ----
__device__ __forceinline__ unsigned fkey(float f) {
  unsigned u = __float_as_uint(f);
  return (u & 0x80000000u) ? ~u : (u | 0x80000000u);
}
__device__ __forceinline__ float fdec(unsigned k) {
  unsigned u = (k & 0x80000000u) ? (k ^ 0x80000000u) : ~k;
  return __uint_as_float(u);
}
__device__ __forceinline__ int refl(int p, int n) {
  if (p < 0) p = -p;
  if (p >= n) p = 2 * n - 2 - p;
  return p;
}
__device__ __forceinline__ float clampf(float v, float lo, float hi) {
  return fminf(fmaxf(v, lo), hi);
}

// ---------------- init ----------------
__global__ void k_init(WsHeader* h) {
  h->xmin_key = 0xFFFFFFFFu; h->xmax_key = 0u;
  h->lmin_key = 0xFFFFFFFFu; h->lmax_key = 0u;
  h->bmin_key = 0xFFFFFFFFu; h->bmax_key = 0u;
  h->sum = 0.0; h->sumsq = 0.0;
}

// ---------------- pass 1: min/max of raw x ----------------
__global__ void k_minmax_x(const float4* __restrict__ x4, WsHeader* h) {
  __shared__ float smn[256], smx[256];
  const unsigned n4 = N_TOT / 4;
  float mn = INFINITY, mx = -INFINITY;
  for (unsigned i = blockIdx.x * blockDim.x + threadIdx.x; i < n4;
       i += gridDim.x * blockDim.x) {
    float4 v = x4[i];
    mn = fminf(mn, fminf(fminf(v.x, v.y), fminf(v.z, v.w)));
    mx = fmaxf(mx, fmaxf(fmaxf(v.x, v.y), fmaxf(v.z, v.w)));
  }
  int tid = threadIdx.x;
  smn[tid] = mn; smx[tid] = mx; __syncthreads();
  for (int s = 128; s > 0; s >>= 1) {
    if (tid < s) { smn[tid] = fminf(smn[tid], smn[tid + s]);
                   smx[tid] = fmaxf(smx[tid], smx[tid + s]); }
    __syncthreads();
  }
  if (tid == 0) { atomicMin(&h->xmin_key, fkey(smn[0]));
                  atomicMax(&h->xmax_key, fkey(smx[0])); }
}

__global__ void k_fin0(WsHeader* h) {
  float mn = fdec(h->xmin_key), mx = fdec(h->xmax_key);
  h->mn = mn;
  h->inv_range = 1.0f / (mx - mn);
}

// ---------------- pass 2: stats of l = clip(log2(1+xn),0,1) ----------------
__global__ void k_stats(const float4* __restrict__ x4, WsHeader* h) {
  __shared__ double ss[256], ss2[256];
  __shared__ float  smn[256], smx[256];
  const float mn = h->mn, ir = h->inv_range;
  const unsigned n4 = N_TOT / 4;
  double s = 0.0, s2 = 0.0;
  float lmn = INFINITY, lmx = -INFINITY;
  for (unsigned i = blockIdx.x * blockDim.x + threadIdx.x; i < n4;
       i += gridDim.x * blockDim.x) {
    float4 v = x4[i];
    float p[4] = {v.x, v.y, v.z, v.w};
#pragma unroll
    for (int k = 0; k < 4; ++k) {
      float xn = (p[k] - mn) * ir;
      float l = clampf(log2f(1.0f + xn), 0.0f, 1.0f);
      s += (double)l; s2 += (double)l * (double)l;
      lmn = fminf(lmn, l); lmx = fmaxf(lmx, l);
    }
  }
  int tid = threadIdx.x;
  ss[tid] = s; ss2[tid] = s2; smn[tid] = lmn; smx[tid] = lmx; __syncthreads();
  for (int st = 128; st > 0; st >>= 1) {
    if (tid < st) {
      ss[tid] += ss[tid + st]; ss2[tid] += ss2[tid + st];
      smn[tid] = fminf(smn[tid], smn[tid + st]);
      smx[tid] = fmaxf(smx[tid], smx[tid + st]);
    }
    __syncthreads();
  }
  if (tid == 0) {
    atomicAdd(&h->sum, ss[0]); atomicAdd(&h->sumsq, ss2[0]);
    atomicMin(&h->lmin_key, fkey(smn[0]));
    atomicMax(&h->lmax_key, fkey(smx[0]));
  }
}

__global__ void k_fin1(WsHeader* h) {
  const double n = (double)N_TOT;
  double mean = h->sum / n;
  double var  = (h->sumsq - h->sum * h->sum / n) / (n - 1.0);   // ddof=1
  float  sd   = (float)sqrt(var);
  float  mf   = (float)mean;
  float lmn = fdec(h->lmin_key), lmx = fdec(h->lmax_key);
  float zmin = (lmn - mf) / sd, zmax = (lmx - mf) / sd;
  float a = 1.0f / (zmax - zmin);
  h->su = a / sd;                    // u = (z - zmin)/(zmax-zmin), z=(l-mean)/sd
  h->ou = (-mf / sd - zmin) * a;
}

// ---------------- pass 3: per-tile histogram + WMMA prefix-sum LUT ----------------
__global__ void k_hist_lut(const float* __restrict__ x, const WsHeader* __restrict__ h,
                           float* __restrict__ lut) {
  __shared__ unsigned shist[NB];
  __shared__ float    sclip[NB];   // clipped counts, H[i][j] = sclip[16i+j]
  __shared__ float    sP[NB];      // within-row inclusive prefix (WMMA result)
  __shared__ float    srow[16];    // per-row totals
  __shared__ unsigned stotal;

  const int tid = threadIdx.x;
  const int t   = blockIdx.x;                  // 0..2047 (B*GH*GW, C=1)
  const int b   = t >> 6;
  const int gi  = (t >> 3) & 7;
  const int gj  = t & 7;
  const float su = h->su, ou = h->ou, mn = h->mn, ir = h->inv_range;

  shist[tid] = 0u;
  if (tid == 0) stotal = 0u;
  __syncthreads();

  // 16384 pixels, float4-vectorized, coalesced
  const size_t base = (size_t)b * (IMG_H * IMG_W) + (size_t)(gi * TILE) * IMG_W + gj * TILE;
#pragma unroll 1
  for (int it = 0; it < 16; ++it) {
    int q = it * 1024 + tid * 4;               // tile-linear pixel index
    int ly = q >> 7, lx = q & 127;
    float4 v = *(const float4*)(x + base + (size_t)ly * IMG_W + lx);
    float p[4] = {v.x, v.y, v.z, v.w};
#pragma unroll
    for (int k = 0; k < 4; ++k) {
      float xn = (p[k] - mn) * ir;
      float l  = clampf(log2f(1.0f + xn), 0.0f, 1.0f);
      float u  = su * l + ou;
      int bin  = (int)(u * 256.0f);
      bin = bin < 0 ? 0 : (bin > 255 ? 255 : bin);
      atomicAdd(&shist[bin], 1u);
    }
  }
  __syncthreads();

  // clip to 192 and accumulate the total of clipped counts
  unsigned c = shist[tid]; if (c > 192u) c = 192u;
  sclip[tid] = (float)c;
  atomicAdd(&stotal, c);
  __syncthreads();

  // ---- WMMA prefix sum: P = H(16x16, padded K->32) x U(upper-tri ones) ----
  // f16 operands are exact (counts <= 192); f32 accumulate is exact.
  {
    const int l  = tid & 31;
    const int m  = l & 15;       // A row / B column / D column
    const int hi = l >> 4;
    v16h A, Bm;
#pragma unroll
    for (int e = 0; e < 16; ++e) {
      int v = e >> 1;
      int K = (v < 4 ? 2 * v : 16 + 2 * (v - 4)) + hi * 8 + (e & 1);
      A[e]  = (_Float16)((K < 16) ? sclip[m * 16 + K] : 0.0f);  // A[m][K]
      int k = (e < 8 ? e : e + 8) + hi * 8;                     // B[k][n], n=m
      Bm[e] = (_Float16)((k <= m && k < 16) ? 1.0f : 0.0f);
    }
    v8f acc = {};
    acc = __builtin_amdgcn_wmma_f32_16x16x32_f16(
        /*neg_a=*/false, A, /*neg_b=*/false, Bm,
        /*c_mod=*/(short)0, acc, /*reuse_a=*/false, /*reuse_b=*/false);
#pragma unroll
    for (int v = 0; v < 8; ++v) {
      int row = v + 8 * hi;                 // D[row][m]
      sP[row * 16 + m] = acc[v];            // waves write identical values
    }
  }
  __syncthreads();
  if (tid < 16) srow[tid] = sP[tid * 16 + 15];   // row totals
  __syncthreads();

  // finish: cross-row offsets + analytic excess redistribution + floored LUT
  int i = tid >> 4;
  float T = 0.0f;
  for (int k = 0; k < i; ++k) T += srow[k];
  float excess = (16384.0f - (float)stotal) * (1.0f / 256.0f);
  float cum = sP[tid] + T + (float)(tid + 1) * excess;
  float lv = floorf(cum * (255.0f / 16384.0f));
  lut[(size_t)t * NB + tid] = clampf(lv, 0.0f, 255.0f);
}

// ---------------- pass 4: fused CLAHE apply + 3x3 gaussian blur ----------------
__device__ __forceinline__ float clahe_val(float xv, float su, float ou, float mn,
                                           float ir, int b, int gy, int gx,
                                           const float* __restrict__ lut) {
  float xn = (xv - mn) * ir;
  float l  = clampf(log2f(1.0f + xn), 0.0f, 1.0f);
  float u  = su * l + ou;
  int bin  = (int)(u * 256.0f);
  bin = bin < 0 ? 0 : (bin > 255 ? 255 : bin);
  float fy = (gy + 0.5f) * (1.0f / TILE) - 0.5f;
  float fx = (gx + 0.5f) * (1.0f / TILE) - 0.5f;
  int i0 = (int)floorf(fy); i0 = i0 < 0 ? 0 : (i0 > 7 ? 7 : i0);
  int j0 = (int)floorf(fx); j0 = j0 < 0 ? 0 : (j0 > 7 ? 7 : j0);
  int i1 = (i0 + 1 > 7) ? 7 : i0 + 1;
  int j1 = (j0 + 1 > 7) ? 7 : j0 + 1;
  float wy = clampf(fy - (float)i0, 0.0f, 1.0f);
  float wx = clampf(fx - (float)j0, 0.0f, 1.0f);
  const float* lb = lut + (size_t)b * 64 * NB;    // L2-resident (2 MB total)
  float v00 = lb[(i0 * 8 + j0) * NB + bin];
  float v01 = lb[(i0 * 8 + j1) * NB + bin];
  float v10 = lb[(i1 * 8 + j0) * NB + bin];
  float v11 = lb[(i1 * 8 + j1) * NB + bin];
  float v = (1.0f - wy) * ((1.0f - wx) * v00 + wx * v01)
          +          wy * ((1.0f - wx) * v10 + wx * v11);
  return v * (1.0f / 255.0f);
}

__global__ void k_apply_blur(const float* __restrict__ x, const float* __restrict__ lut,
                             WsHeader* h, float* __restrict__ out) {
  __shared__ float sv[34 * 34];
  __shared__ float smn[256], smx[256];
  const int tid = threadIdx.x;
  const int tx0 = blockIdx.x * 32;
  const int ty0 = blockIdx.y * 32;
  const int b   = blockIdx.z;
  const float su = h->su, ou = h->ou, mn = h->mn, ir = h->inv_range;

  for (int it = 0; it < 5; ++it) {
    int idx = it * 256 + tid;
    if (idx < 34 * 34) {
      int hy = idx / 34, hx = idx - hy * 34;
      int gy = refl(ty0 - 1 + hy, IMG_H);
      int gx = refl(tx0 - 1 + hx, IMG_W);
      float xv = x[(size_t)b * (IMG_H * IMG_W) + (size_t)gy * IMG_W + gx];
      sv[idx] = clahe_val(xv, su, ou, mn, ir, b, gy, gx, lut);
    }
  }
  __syncthreads();

  // separable 3x3 gaussian, sigma=1: k = [a,1,a]/(1+2a), a = exp(-0.5)
  const float a  = 0.60653065971263342f;
  const float kb = 1.0f / (1.0f + 2.0f * a);
  const float ka = a * kb;
  const float w[3] = {ka, kb, ka};
  const int ox  = tid & 31;
  const int oy0 = (tid >> 5) * 4;
  float bmn = INFINITY, bmx = -INFINITY;
#pragma unroll
  for (int r = 0; r < 4; ++r) {
    int oy = oy0 + r;
    float s = 0.0f;
#pragma unroll
    for (int di = 0; di < 3; ++di)
#pragma unroll
      for (int dj = 0; dj < 3; ++dj)
        s += w[di] * w[dj] * sv[(oy + di) * 34 + (ox + dj)];
    out[(size_t)b * (IMG_H * IMG_W) + (size_t)(ty0 + oy) * IMG_W + (tx0 + ox)] = s;
    bmn = fminf(bmn, s); bmx = fmaxf(bmx, s);
  }
  smn[tid] = bmn; smx[tid] = bmx; __syncthreads();
  for (int st = 128; st > 0; st >>= 1) {
    if (tid < st) { smn[tid] = fminf(smn[tid], smn[tid + st]);
                    smx[tid] = fmaxf(smx[tid], smx[tid + st]); }
    __syncthreads();
  }
  if (tid == 0) { atomicMin(&h->bmin_key, fkey(smn[0]));
                  atomicMax(&h->bmax_key, fkey(smx[0])); }
}

// ---------------- pass 5: final normalize (in place) + y passthrough ----------------
__global__ void k_final(float* __restrict__ out, const WsHeader* __restrict__ h) {
  float bmn = fdec(h->bmin_key);
  float inv = 1.0f / (fdec(h->bmax_key) - bmn);
  for (unsigned i = blockIdx.x * blockDim.x + threadIdx.x; i < N_TOT;
       i += gridDim.x * blockDim.x)
    out[i] = (out[i] - bmn) * inv;
}

__global__ void k_copy_y(const int* __restrict__ y, int* __restrict__ out2) {
  for (unsigned i = blockIdx.x * blockDim.x + threadIdx.x; i < N_TOT;
       i += gridDim.x * blockDim.x)
    out2[i] = y[i];
}

extern "C" void kernel_launch(void* const* d_in, const int* in_sizes, int n_in,
                              void* d_out, int out_size, void* d_ws, size_t ws_size,
                              hipStream_t stream) {
  (void)in_sizes; (void)n_in; (void)out_size; (void)ws_size;
  const float* x = (const float*)d_in[0];
  const int*   y = (const int*)d_in[1];
  float* out = (float*)d_out;
  WsHeader* h = (WsHeader*)d_ws;
  float* lut = (float*)((char*)d_ws + 256);   // 2048*256 floats = 2 MB

  k_init<<<1, 1, 0, stream>>>(h);
  k_minmax_x<<<2048, 256, 0, stream>>>((const float4*)x, h);
  k_fin0<<<1, 1, 0, stream>>>(h);
  k_stats<<<2048, 256, 0, stream>>>((const float4*)x, h);
  k_fin1<<<1, 1, 0, stream>>>(h);
  k_hist_lut<<<2048, 256, 0, stream>>>(x, h, lut);
  dim3 g4(IMG_W / 32, IMG_H / 32, IMG_B);
  k_apply_blur<<<g4, 256, 0, stream>>>(x, lut, h, out);
  k_final<<<8192, 256, 0, stream>>>(out, h);
  k_copy_y<<<8192, 256, 0, stream>>>(y, (int*)(out + N_TOT));
}
